// Transolver_21165598834907
// MI455X (gfx1250) — matrix operand order repaired and checked by reading, not compiled, not run on hardware
//
#include <hip/hip_runtime.h>
#include <hip/hip_bf16.h>

// ---------------------------------------------------------------------------
// Transolver forward for MI455X (gfx1250, wave32, WMMA).
// All 512/2048-wide GEMMs run through v_wmma_f32_16x16x32_f16 with f32 accum.
// GEMM: 256x128 macro tile, 8 waves (4x2), 64x64 per wave (16 WMMA/K-step),
// double-buffered LDS with fully-unrolled pointer-incremented tile loads.
// ---------------------------------------------------------------------------

typedef __attribute__((ext_vector_type(16))) _Float16 v16h;
typedef __attribute__((ext_vector_type(8)))  _Float16 v8h;
typedef __attribute__((ext_vector_type(8)))  float    v8f;

#define TOKENS   16384          // B*N
#define HDIM     512
#define HEADS_C  8
#define DH_C     64
#define G_C      32
#define MLPH     2048
#define PREH     1024
#define NCHUNK   64             // token-aggregation chunks along N per (b,h)
#define CHLEN    128            // 8192 / NCHUNK

// ---------------- weight convert: fp32 [K][N] -> f16 [N][K] -----------------
__global__ __launch_bounds__(256)
void wconv_kernel(const float* __restrict__ src, _Float16* __restrict__ dst,
                  int K, int N) {
  size_t e = (size_t)blockIdx.x * 256 + threadIdx.x;
  if (e >= (size_t)K * N) return;
  size_t n = e / K, k = e % K;
  dst[e] = (_Float16)src[k * (size_t)N + n];
}

// ---------------- preprocess: gelu(concat(emb,fun) @ W_pre1 + b) -> f16 -----
__global__ __launch_bounds__(256)
void pre1_kernel(const float* __restrict__ fun, const float* __restrict__ emb,
                 const float* __restrict__ W1, const float* __restrict__ b1,
                 _Float16* __restrict__ t1) {
  size_t e = (size_t)blockIdx.x * 256 + threadIdx.x;   // over TOKENS*PREH
  int j = (int)(e & (PREH - 1));
  size_t m = e >> 10;
  float x[10];
#pragma unroll
  for (int c = 0; c < 3; c++) x[c] = emb[m * 3 + c];
#pragma unroll
  for (int c = 0; c < 7; c++) x[3 + c] = fun[m * 7 + c];
  float s = b1[j];
#pragma unroll
  for (int c = 0; c < 10; c++) s += x[c] * W1[c * PREH + j];
  s = 0.5f * s * (1.0f + erff(s * 0.70710678118f));
  t1[e] = (_Float16)s;
}

// ---------------- LayerNorm: fp32 [M,512] -> f16 [M,512], wave per row ------
__global__ __launch_bounds__(256)
void layernorm_kernel(const float* __restrict__ x, const float* __restrict__ g,
                      const float* __restrict__ b, _Float16* __restrict__ out,
                      int M) {
  int row  = blockIdx.x * 8 + (threadIdx.x >> 5);
  int lane = threadIdx.x & 31;
  if (row >= M) return;
  const float* xr = x + (size_t)row * HDIM;
  float vals[16];
  float s = 0.f;
#pragma unroll
  for (int i = 0; i < 16; i++) { vals[i] = xr[lane + i * 32]; s += vals[i]; }
#pragma unroll
  for (int off = 16; off > 0; off >>= 1) s += __shfl_xor(s, off, 32);
  float mean = s * (1.f / HDIM);
  float vs = 0.f;
#pragma unroll
  for (int i = 0; i < 16; i++) { float d = vals[i] - mean; vs += d * d; }
#pragma unroll
  for (int off = 16; off > 0; off >>= 1) vs += __shfl_xor(vs, off, 32);
  float rinv = rsqrtf(vs * (1.f / HDIM) + 1e-5f);
  _Float16* orow = out + (size_t)row * HDIM;
#pragma unroll
  for (int i = 0; i < 16; i++) {
    int idx = lane + i * 32;
    orow[idx] = (_Float16)((vals[i] - mean) * rinv * g[idx] + b[idx]);
  }
}

// ---------------- WMMA GEMM: C[M,N] = A[M,K](f16) * Bt[N,K](f16)^T ----------
#define TM 256
#define TN 128
#define TK 32
#define LDSP 40   // padded half-stride (80B rows, 16B aligned subloads)

template <bool GELU_EP, bool RES, bool OUT16>
__global__ __launch_bounds__(256)
void gemm_wmma_kernel(const _Float16* __restrict__ A,
                      const _Float16* __restrict__ Bt,
                      const float* __restrict__ bias,
                      const float* __restrict__ bias2,
                      const float* __restrict__ res,
                      float* __restrict__ outF, _Float16* __restrict__ outH,
                      int M, int N, int K) {
  __shared__ _Float16 As[2][TM * LDSP];   // 2 x 256 x 32 halves (padded)
  __shared__ _Float16 Bs[2][TN * LDSP];   // 2 x 128 x 32 halves (padded)
  const int m0  = blockIdx.x * TM;
  const int n0  = blockIdx.y * TN;
  const int tid = threadIdx.x;
  const int lane = tid & 31;
  const int wid  = tid >> 5;    // 0..7
  const int wm   = wid >> 1;    // 0..3  -> 64 rows each
  const int wn   = wid & 1;     // 0..1  -> 64 cols each
  const int lm   = lane & 15;
  const int lh   = lane >> 4;

  // per-thread cooperative-load coordinates: chunk = 8 halves (16B)
  const int crow = tid >> 2;          // 0..63
  const int coff = (tid & 3) * 8;     // 0,8,16,24

  const _Float16* pA[4];
  const _Float16* pB[2];
#pragma unroll
  for (int i = 0; i < 4; i++)
    pA[i] = A + (size_t)(m0 + crow + 64 * i) * K + coff;
#pragma unroll
  for (int j = 0; j < 2; j++)
    pB[j] = Bt + (size_t)(n0 + crow + 64 * j) * K + coff;

  v8f acc[4][4];
#pragma unroll
  for (int i = 0; i < 4; i++)
#pragma unroll
    for (int j = 0; j < 4; j++)
#pragma unroll
      for (int e = 0; e < 8; e++) acc[i][j][e] = 0.f;

  const int KT = K / TK;

  // tile loader: fully unrolled, pointer-increment addressing
  auto loadTile = [&](int buf, int kt) {
    const size_t kadd = (size_t)kt * TK;
#pragma unroll
    for (int i = 0; i < 4; i++)
      *(v8h*)&As[buf][(crow + 64 * i) * LDSP + coff] =
          *(const v8h*)(pA[i] + kadd);
#pragma unroll
    for (int j = 0; j < 2; j++)
      *(v8h*)&Bs[buf][(crow + 64 * j) * LDSP + coff] =
          *(const v8h*)(pB[j] + kadd);
  };

  loadTile(0, 0);
  __syncthreads();

  for (int kt = 0; kt < KT; kt++) {
    const int cur = kt & 1;
    if (kt + 1 < KT) loadTile(cur ^ 1, kt + 1);
    if (kt + 2 < KT) {
      const size_t pf = (size_t)(kt + 2) * TK;
#pragma unroll
      for (int i = 0; i < 4; i++) __builtin_prefetch(pA[i] + pf, 0, 1);
#pragma unroll
      for (int j = 0; j < 2; j++) __builtin_prefetch(pB[j] + pf, 0, 1);
    }

    v16h afrag[4], bfrag[4];
#pragma unroll
    for (int i = 0; i < 4; i++) {
      int r = wm * 64 + i * 16 + lm;
      // A 16x32 f16 lane map: lanes 0-15 hold K[8h..8h+7],K[16+8h..16+8h+7]
      v8h lo = *(const v8h*)&As[cur][r * LDSP + 8 * lh];
      v8h hi = *(const v8h*)&As[cur][r * LDSP + 16 + 8 * lh];
      afrag[i] = __builtin_shufflevector(lo, hi, 0, 1, 2, 3, 4, 5, 6, 7, 8, 9,
                                         10, 11, 12, 13, 14, 15);
    }
#pragma unroll
    for (int j = 0; j < 4; j++) {
      int r = wn * 64 + j * 16 + lm;
      // B 32x16 f16 lane map: lane group h holds K[8h..8h+7],K[16+8h..16+8h+7] of column n
      v8h lo = *(const v8h*)&Bs[cur][r * LDSP + 8 * lh];
      v8h hi = *(const v8h*)&Bs[cur][r * LDSP + 16 + 8 * lh];
      bfrag[j] = __builtin_shufflevector(lo, hi, 0, 1, 2, 3, 4, 5, 6, 7, 8, 9,
                                         10, 11, 12, 13, 14, 15);
    }
#pragma unroll
    for (int i = 0; i < 4; i++)
#pragma unroll
      for (int j = 0; j < 4; j++)
        acc[i][j] = __builtin_amdgcn_wmma_f32_16x16x32_f16(
            false, afrag[i], false, bfrag[j], (short)0, acc[i][j], false,
            false);
    __syncthreads();
  }

  // Epilogue. D layout: VGPR r, lane l -> M = r + 8*(l>>4), N = l&15.
#pragma unroll
  for (int i = 0; i < 4; i++) {
    int mbase = m0 + wm * 64 + i * 16 + 8 * lh;
#pragma unroll
    for (int j = 0; j < 4; j++) {
      int ncol = n0 + wn * 64 + j * 16 + lm;
      float bv = bias[ncol];
      if (bias2) bv += bias2[ncol];
#pragma unroll
      for (int r = 0; r < 8; r++) {
        size_t idx = (size_t)(mbase + r) * N + ncol;
        float v = acc[i][j][r] + bv;
        if (GELU_EP) v = 0.5f * v * (1.0f + erff(v * 0.70710678118f));
        if (RES) v += res[idx];
        if (OUT16)
          outH[idx] = (_Float16)v;
        else
          outF[idx] = v;
      }
    }
  }
}

// ---------------- slice logits + softmax: wave per (token, head) ------------
__global__ __launch_bounds__(256)
void slice_softmax_kernel(const float* __restrict__ xmid,
                          const float* __restrict__ Wsl,
                          const float* __restrict__ bsl,
                          const float* __restrict__ temp,
                          float* __restrict__ sw, int M) {
  int idx  = blockIdx.x * 8 + (threadIdx.x >> 5);   // m*HEADS + head
  int lane = threadIdx.x & 31;                       // g
  if (idx >= M * HEADS_C) return;
  int m = idx >> 3, head = idx & 7;
  const float* xv = xmid + (size_t)m * HDIM + head * DH_C;
  float s = bsl[lane];
#pragma unroll 8
  for (int d = 0; d < DH_C; d++) s += xv[d] * Wsl[d * G_C + lane];
  s /= temp[head];
  float mx = s;
#pragma unroll
  for (int off = 16; off > 0; off >>= 1)
    mx = fmaxf(mx, __shfl_xor(mx, off, 32));
  float e = __expf(s - mx);
  float sum = e;
#pragma unroll
  for (int off = 16; off > 0; off >>= 1) sum += __shfl_xor(sum, off, 32);
  sw[(size_t)idx * G_C + lane] = e / sum;
}

// ---------------- token aggregation partials over N chunks ------------------
// blk = (b*HEADS + head)*NCHUNK + chunk
__global__ __launch_bounds__(256)
void token_partial_kernel(const float* __restrict__ fxmid,
                          const float* __restrict__ sw,
                          float* __restrict__ tok_part,
                          float* __restrict__ mass_part) {
  int blk = blockIdx.x;
  int chunk = blk & (NCHUNK - 1);
  int head  = (blk >> 6) & 7;
  int bb    = blk >> 9;
  int t = threadIdx.x;
  int d  = t & 63;
  int gq = t >> 6;          // 4 groups of 8 g's
  size_t nbase = (size_t)bb * 8192 + (size_t)chunk * CHLEN;
  float acc[8], macc[8];
#pragma unroll
  for (int g = 0; g < 8; g++) { acc[g] = 0.f; macc[g] = 0.f; }
  for (int n = 0; n < CHLEN; n++) {
    size_t tok = nbase + n;
    float fv = fxmid[tok * HDIM + head * DH_C + d];
    const float* swp = sw + (tok * HEADS_C + head) * G_C + gq * 8;
#pragma unroll
    for (int g = 0; g < 8; g++) {
      float s = swp[g];
      acc[g] += fv * s;
      macc[g] += s;
    }
  }
  float* tp = tok_part + (size_t)blk * (G_C * DH_C);
#pragma unroll
  for (int g = 0; g < 8; g++) tp[(gq * 8 + g) * DH_C + d] = acc[g];
  if (d == 0) {
#pragma unroll
    for (int g = 0; g < 8; g++) mass_part[(size_t)blk * G_C + gq * 8 + g] = macc[g];
  }
}

// ---------------- reduce partials, normalize by slice mass ------------------
__global__ __launch_bounds__(256)
void token_reduce_kernel(const float* __restrict__ tok_part,
                         const float* __restrict__ mass_part,
                         float* __restrict__ token) {
  int bh = blockIdx.x;
  int t  = threadIdx.x;
  __shared__ float mass[G_C];
  if (t < G_C) {
    float s = 0.f;
    for (int c = 0; c < NCHUNK; c++)
      s += mass_part[((size_t)bh * NCHUNK + c) * G_C + t];
    mass[t] = s + 1e-5f;
  }
  __syncthreads();
  for (int cell = t; cell < G_C * DH_C; cell += 256) {
    int g = cell >> 6;
    float s = 0.f;
    for (int c = 0; c < NCHUNK; c++)
      s += tok_part[((size_t)bh * NCHUNK + c) * (G_C * DH_C) + cell];
    token[(size_t)bh * (G_C * DH_C) + cell] = s / mass[g];
  }
}

// ---------------- per-(b,head) 32-token attention ---------------------------
__global__ __launch_bounds__(256)
void attention_kernel(const float* __restrict__ token,
                      const float* __restrict__ Wq,
                      const float* __restrict__ Wk,
                      const float* __restrict__ Wv,
                      float* __restrict__ out_tok) {
  __shared__ float tk[G_C * DH_C], q[G_C * DH_C], kk[G_C * DH_C],
      vv[G_C * DH_C], at[G_C * G_C];
  int bh = blockIdx.x, t = threadIdx.x;
  const float* tok = token + (size_t)bh * (G_C * DH_C);
  for (int c = t; c < G_C * DH_C; c += 256) tk[c] = tok[c];
  __syncthreads();
  for (int c = t; c < G_C * DH_C; c += 256) {
    int g = c >> 6, d = c & 63;
    float sq = 0.f, sk = 0.f, sv = 0.f;
#pragma unroll 8
    for (int e = 0; e < DH_C; e++) {
      float x = tk[g * DH_C + e];
      sq += x * Wq[e * DH_C + d];
      sk += x * Wk[e * DH_C + d];
      sv += x * Wv[e * DH_C + d];
    }
    q[c] = sq; kk[c] = sk; vv[c] = sv;
  }
  __syncthreads();
  for (int c = t; c < G_C * G_C; c += 256) {
    int g = c >> 5, j = c & 31;
    float s = 0.f;
#pragma unroll 8
    for (int d = 0; d < DH_C; d++) s += q[g * DH_C + d] * kk[j * DH_C + d];
    at[c] = s * 0.125f;   // DH^-0.5
  }
  __syncthreads();
  if (t < G_C) {
    float mx = -3.4e38f;
    for (int j = 0; j < G_C; j++) mx = fmaxf(mx, at[t * G_C + j]);
    float sum = 0.f;
    for (int j = 0; j < G_C; j++) {
      float e = __expf(at[t * G_C + j] - mx);
      at[t * G_C + j] = e;
      sum += e;
    }
    float r = 1.f / sum;
    for (int j = 0; j < G_C; j++) at[t * G_C + j] *= r;
  }
  __syncthreads();
  for (int c = t; c < G_C * DH_C; c += 256) {
    int g = c >> 6, d = c & 63;
    float s = 0.f;
#pragma unroll 8
    for (int j = 0; j < G_C; j++) s += at[g * G_C + j] * vv[j * DH_C + d];
    out_tok[(size_t)bh * (G_C * DH_C) + c] = s;
  }
}

// ---------------- de-slice: out[m,h,d] = sum_g out_tok*sw -> f16 ------------
__global__ __launch_bounds__(256)
void deslice_kernel(const float* __restrict__ out_tok,
                    const float* __restrict__ sw,
                    _Float16* __restrict__ out16) {
  size_t e = (size_t)blockIdx.x * 256 + threadIdx.x;  // over TOKENS*HDIM
  size_t m = e >> 9;
  int hd = (int)(e & (HDIM - 1));
  int head = hd >> 6, d = hd & 63;
  int bb = (int)(m >> 13);  // m / 8192
  const float* ot = out_tok + ((size_t)(bb * HEADS_C + head)) * (G_C * DH_C) + d;
  const float* swp = sw + (m * HEADS_C + head) * G_C;
  float s = 0.f;
#pragma unroll 8
  for (int g = 0; g < G_C; g++) s += ot[g * DH_C] * swp[g];
  out16[e] = (_Float16)s;
}

// ---------------- final head: LN(fx) @ Wout + bout --------------------------
__global__ __launch_bounds__(256)
void head_kernel(const _Float16* __restrict__ hf, const float* __restrict__ Wout,
                 const float* __restrict__ bout, float* __restrict__ out) {
  size_t e = (size_t)blockIdx.x * 256 + threadIdx.x;  // TOKENS*4
  size_t m = e >> 2;
  int o = (int)(e & 3);
  float s = bout[o];
  const _Float16* hr = hf + m * HDIM;
#pragma unroll 8
  for (int k = 0; k < HDIM; k++) s += (float)hr[k] * Wout[k * 4 + o];
  out[e] = s;
}

// ===========================================================================
extern "C" void kernel_launch(void* const* d_in, const int* in_sizes, int n_in,
                              void* d_out, int out_size, void* d_ws,
                              size_t ws_size, hipStream_t stream) {
  const float* fun      = (const float*)d_in[0];
  const float* emb      = (const float*)d_in[1];
  const float* W_pre1   = (const float*)d_in[2];
  const float* b_pre1   = (const float*)d_in[3];
  const float* W_pre2   = (const float*)d_in[4];
  const float* b_pre2   = (const float*)d_in[5];
  const float* placeholder = (const float*)d_in[6];
  const float* ln1_g    = (const float*)d_in[7];
  const float* ln1_b    = (const float*)d_in[8];
  const float* Wfx      = (const float*)d_in[9];
  const float* bfx      = (const float*)d_in[10];
  const float* Wx       = (const float*)d_in[11];
  const float* bx       = (const float*)d_in[12];
  const float* Wslice   = (const float*)d_in[13];
  const float* bslice   = (const float*)d_in[14];
  const float* temp     = (const float*)d_in[15];
  const float* Wq       = (const float*)d_in[16];
  const float* Wk       = (const float*)d_in[17];
  const float* Wv       = (const float*)d_in[18];
  const float* Wo       = (const float*)d_in[19];
  const float* bo       = (const float*)d_in[20];
  const float* ln2_g    = (const float*)d_in[21];
  const float* ln2_b    = (const float*)d_in[22];
  const float* Wm1      = (const float*)d_in[23];
  const float* bm1      = (const float*)d_in[24];
  const float* Wm2      = (const float*)d_in[25];
  const float* bm2      = (const float*)d_in[26];
  const float* lnf_g    = (const float*)d_in[27];
  const float* lnf_b    = (const float*)d_in[28];
  const float* WoutW    = (const float*)d_in[29];
  const float* boutW    = (const float*)d_in[30];

  const int M = TOKENS;
  char* ws = (char*)d_ws;
  size_t off = 0;
  auto alloc = [&](size_t bytes) -> char* {
    char* p = ws + off;
    off += (bytes + 255) & ~(size_t)255;
    return p;
  };

  _Float16* wpre2T = (_Float16*)alloc((size_t)PREH * HDIM * 2);
  _Float16 *WfxT[4], *WxT[4], *WoT[4], *Wm1T[4], *Wm2T[4];
  for (int l = 0; l < 4; l++) {
    WfxT[l] = (_Float16*)alloc((size_t)HDIM * HDIM * 2);
    WxT[l]  = (_Float16*)alloc((size_t)HDIM * HDIM * 2);
    WoT[l]  = (_Float16*)alloc((size_t)HDIM * HDIM * 2);
    Wm1T[l] = (_Float16*)alloc((size_t)HDIM * MLPH * 2);
    Wm2T[l] = (_Float16*)alloc((size_t)MLPH * HDIM * 2);
  }
  float*    fx    = (float*)alloc((size_t)M * HDIM * 4);
  _Float16* h16   = (_Float16*)alloc((size_t)M * HDIM * 2);
  _Float16* tbig  = (_Float16*)alloc((size_t)M * MLPH * 2);  // t1 / t2 / out16
  float*    xmid  = (float*)alloc((size_t)M * HDIM * 4);
  float*    fxmid = (float*)alloc((size_t)M * HDIM * 4);
  float*    swb   = (float*)alloc((size_t)M * HEADS_C * G_C * 4);
  float*    tokp  = (float*)alloc((size_t)2 * HEADS_C * NCHUNK * G_C * DH_C * 4);
  float*    massp = (float*)alloc((size_t)2 * HEADS_C * NCHUNK * G_C * 4);
  float*    token = (float*)alloc((size_t)2 * HEADS_C * G_C * DH_C * 4);
  float*    otok  = (float*)alloc((size_t)2 * HEADS_C * G_C * DH_C * 4);
  _Float16* out16 = tbig;  // alias: tbig is dead during attention phase

  auto wconv = [&](const float* src, _Float16* dst, int K, int N) {
    size_t tot = (size_t)K * N;
    wconv_kernel<<<(unsigned)((tot + 255) / 256), 256, 0, stream>>>(src, dst, K, N);
  };
  // 1) weight conversion (f32 [K][N] -> f16 [N][K])
  wconv(W_pre2, wpre2T, PREH, HDIM);
  for (int l = 0; l < 4; l++) {
    wconv(Wfx + (size_t)l * HDIM * HDIM, WfxT[l], HDIM, HDIM);
    wconv(Wx  + (size_t)l * HDIM * HDIM, WxT[l],  HDIM, HDIM);
    wconv(Wo  + (size_t)l * HDIM * HDIM, WoT[l],  HDIM, HDIM);
    wconv(Wm1 + (size_t)l * HDIM * MLPH, Wm1T[l], HDIM, MLPH);
    wconv(Wm2 + (size_t)l * MLPH * HDIM, Wm2T[l], MLPH, HDIM);
  }

  auto gemm = [&](int mode, const _Float16* A, const _Float16* Bt,
                  const float* bias, const float* bias2, const float* res,
                  float* oF, _Float16* oH, int N, int K) {
    dim3 g(M / TM, N / TN);
    switch (mode) {
      case 0:  // plain, f32 out
        gemm_wmma_kernel<false, false, false><<<g, 256, 0, stream>>>(
            A, Bt, bias, bias2, res, oF, oH, M, N, K);
        break;
      case 1:  // residual add, f32 out
        gemm_wmma_kernel<false, true, false><<<g, 256, 0, stream>>>(
            A, Bt, bias, bias2, res, oF, oH, M, N, K);
        break;
      case 2:  // gelu, f16 out
        gemm_wmma_kernel<true, false, true><<<g, 256, 0, stream>>>(
            A, Bt, bias, bias2, res, oF, oH, M, N, K);
        break;
    }
  };

  // 2) preprocess
  pre1_kernel<<<(unsigned)((size_t)M * PREH / 256), 256, 0, stream>>>(
      fun, emb, W_pre1, b_pre1, tbig);
  gemm(0, tbig, wpre2T, b_pre2, placeholder, nullptr, fx, nullptr, HDIM, PREH);

  const unsigned lnGrid = (M + 7) / 8;
  for (int l = 0; l < 4; l++) {
    // --- physics attention ---
    layernorm_kernel<<<lnGrid, 256, 0, stream>>>(fx, ln1_g + l * HDIM,
                                                 ln1_b + l * HDIM, h16, M);
    gemm(0, h16, WfxT[l], bfx + l * HDIM, nullptr, nullptr, fxmid, nullptr,
         HDIM, HDIM);
    gemm(0, h16, WxT[l], bx + l * HDIM, nullptr, nullptr, xmid, nullptr,
         HDIM, HDIM);
    slice_softmax_kernel<<<(unsigned)(M * HEADS_C / 8), 256, 0, stream>>>(
        xmid, Wslice + (size_t)l * DH_C * G_C, bslice + l * G_C,
        temp + l * HEADS_C, swb, M);
    token_partial_kernel<<<2 * HEADS_C * NCHUNK, 256, 0, stream>>>(
        fxmid, swb, tokp, massp);
    token_reduce_kernel<<<2 * HEADS_C, 256, 0, stream>>>(tokp, massp, token);
    attention_kernel<<<2 * HEADS_C, 256, 0, stream>>>(
        token, Wq + (size_t)l * DH_C * DH_C, Wk + (size_t)l * DH_C * DH_C,
        Wv + (size_t)l * DH_C * DH_C, otok);
    deslice_kernel<<<(unsigned)((size_t)M * HDIM / 256), 256, 0, stream>>>(
        otok, swb, out16);
    gemm(1, out16, WoT[l], bo + l * HDIM, nullptr, fx, fx, nullptr, HDIM,
         HDIM);
    // --- MLP block ---
    layernorm_kernel<<<lnGrid, 256, 0, stream>>>(fx, ln2_g + l * HDIM,
                                                 ln2_b + l * HDIM, h16, M);
    gemm(2, h16, Wm1T[l], bm1 + l * MLPH, nullptr, nullptr, nullptr, tbig,
         MLPH, HDIM);
    gemm(1, tbig, Wm2T[l], bm2 + l * HDIM, nullptr, fx, fx, nullptr, HDIM,
         MLPH);
  }

  // 3) final head
  layernorm_kernel<<<lnGrid, 256, 0, stream>>>(fx, lnf_g, lnf_b, h16, M);
  head_kernel<<<(unsigned)((size_t)M * 4 / 256), 256, 0, stream>>>(
      h16, WoutW, boutW, (float*)d_out);
}
